// ConvSNN2_31198642438228
// MI455X (gfx1250) — matrix operand order repaired and checked by reading, not compiled, not run on hardware
//
#include <hip/hip_runtime.h>

typedef __attribute__((ext_vector_type(16))) _Float16 v16h;
typedef __attribute__((ext_vector_type(8)))  _Float16 v8h;
typedef __attribute__((ext_vector_type(8)))  float    v8f;

// ---- model dims ----
#define BATCH 256
#define TSTEPS 40
#define S0   (BATCH*20*26*26)     // conv1/LIF0 elements
#define SP   (BATCH*20*13*13)     // pooled elements
#define M2   (BATCH*121)          // conv2 implicit-GEMM rows (30976)
#define K2P  192                  // 20*9=180 padded to 192
#define N2P  64                   // 50 padded to 64
#define S1   (BATCH*50*121)       // LIF1 elements (6050/batch)
#define KF1  6050
#define KF1P 6080                 // padded K (multiple of 32)
#define NF1  500
#define NF1P 512
#define KF2P 512                  // 500 padded
#define NF2  200
#define NF2P 224

// ---------------- utility kernels ----------------
__global__ void zero_ws(float* p, size_t n) {
  size_t i = (size_t)blockIdx.x * blockDim.x + threadIdx.x;
  size_t st = (size_t)gridDim.x * blockDim.x;
  for (; i < n; i += st) p[i] = 0.f;
}

// convert f32 weights [rows][krow] row-major -> f16 padded [rows][ld]
__global__ void cvt_w_f16(const float* __restrict__ src, _Float16* __restrict__ dst,
                          int n, int krow, int ld) {
  int idx = blockIdx.x * blockDim.x + threadIdx.x;
  if (idx >= n) return;
  int k = idx % krow, r = idx / krow;
  dst[(size_t)r * ld + k] = (_Float16)src[idx];
}

// ---------------- fused conv1 + LIF0 ----------------
__global__ void conv1_lif0(const float* __restrict__ xt, const float* __restrict__ wc1,
                           float* __restrict__ v0, float* __restrict__ i0,
                           float* __restrict__ z0) {
  int idx = blockIdx.x * blockDim.x + threadIdx.x;
  if (idx >= S0) return;
  int ox = idx % 26; int t1 = idx / 26;
  int oy = t1 % 26;  int t2 = t1 / 26;
  int c  = t2 % 20;  int b  = t2 / 20;
  const float* xp = xt + (size_t)b * 784 + oy * 28 + ox;
  const float* wp = wc1 + c * 9;
  float acc = 0.f;
#pragma unroll
  for (int r = 0; r < 3; ++r)
#pragma unroll
    for (int s = 0; s < 3; ++s)
      acc += xp[r * 28 + s] * wp[r * 3 + s];
  float v = v0[idx], cur = i0[idx];
  float vdec = v + 0.1f * (cur - v);
  float z = (vdec - 0.2f) > 0.f ? 1.f : 0.f;
  v0[idx] = (1.f - z) * vdec;
  i0[idx] = 0.8f * cur + acc;
  z0[idx] = z;
}

// ---------------- 2x2 avg pool -> f16 ----------------
__global__ void pool2(const float* __restrict__ z0, _Float16* __restrict__ ph) {
  int idx = blockIdx.x * blockDim.x + threadIdx.x;
  if (idx >= SP) return;
  int px = idx % 13; int t = idx / 13;
  int py = t % 13;   t /= 13;
  int c = t % 20;    int b = t / 20;
  const float* zp = z0 + (((size_t)(b * 20 + c) * 26 + py * 2) * 26 + px * 2);
  ph[idx] = (_Float16)(0.25f * (zp[0] + zp[1] + zp[26] + zp[27]));
}

// ---------------- im2col for conv2 (k-padded with zeros) ----------------
__global__ void im2col2(const _Float16* __restrict__ ph, _Float16* __restrict__ col) {
  int idx = blockIdx.x * blockDim.x + threadIdx.x;
  if (idx >= M2 * K2P) return;
  int k = idx % K2P; int m = idx / K2P;
  int pos = m % 121; int b = m / 121;
  int oy = pos / 11, ox = pos % 11;
  _Float16 v = (_Float16)0.f;
  if (k < 180) {
    int c = k / 9; int rs = k % 9; int r = rs / 3, s = rs % 3;
    v = ph[((size_t)(b * 20 + c) * 13 + oy + r) * 13 + ox + s];
  }
  col[idx] = v;
}

// ---------------- generic f16 WMMA GEMM: C[m,n] = sum_k A[m,k]*Bw[n,k] ----------------
// Each wave computes a 32x32 tile (2x2 of 16x16x32 WMMA frags).
__global__ __launch_bounds__(256) void gemm_wmma_2x2(
    const _Float16* __restrict__ A, const _Float16* __restrict__ Bw,
    float* __restrict__ C, int M, int Nreal, int ldc,
    int K, int lda, int ldb, int nTiles) {
  int wave = (int)((blockIdx.x * blockDim.x + threadIdx.x) >> 5);
  int lane = threadIdx.x & 31;
  int mTiles = M >> 5;
  if (wave >= mTiles * nTiles) return;
  int mt = wave / nTiles, nt = wave % nTiles;
  int m0 = mt << 5, n0 = nt << 5;
  int lo16 = lane & 15, hi = lane >> 4;

  // per-lane base pointers (ISA 16-bit A / B fragment layouts)
  const _Float16* a0p = A + (size_t)(m0 + lo16) * lda + hi * 8;
  const _Float16* a1p = a0p + (size_t)16 * lda;
  const _Float16* b0p = Bw + (size_t)(n0 + lo16) * ldb + hi * 8;
  const _Float16* b1p = b0p + (size_t)16 * ldb;

  v8f c00 = {}, c01 = {}, c10 = {}, c11 = {};
  for (int k = 0; k < K; k += 32) {
    // WGP-scope prefetch (locality 3): prewarm the near cache for the next
    // chunks; everything is L2-resident, SYS-scope prefetch would be a no-op.
    __builtin_prefetch((const void*)(a0p + k + 128), 0, 3);
    __builtin_prefetch((const void*)(b0p + k + 128), 0, 3);
    v8h a0l = *(const v8h*)(a0p + k);
    v8h a0h = *(const v8h*)(a0p + k + 16);
    v8h a1l = *(const v8h*)(a1p + k);
    v8h a1h = *(const v8h*)(a1p + k + 16);
    v16h a0 = __builtin_shufflevector(a0l, a0h, 0,1,2,3,4,5,6,7,8,9,10,11,12,13,14,15);
    v16h a1 = __builtin_shufflevector(a1l, a1h, 0,1,2,3,4,5,6,7,8,9,10,11,12,13,14,15);
    v16h b0 = __builtin_shufflevector(*(const v8h*)(b0p + k), *(const v8h*)(b0p + k + 16), 0,1,2,3,4,5,6,7,8,9,10,11,12,13,14,15);
    v16h b1 = __builtin_shufflevector(*(const v8h*)(b1p + k), *(const v8h*)(b1p + k + 16), 0,1,2,3,4,5,6,7,8,9,10,11,12,13,14,15);
    c00 = __builtin_amdgcn_wmma_f32_16x16x32_f16(false, a0, false, b0, (short)0, c00, false, false);
    c01 = __builtin_amdgcn_wmma_f32_16x16x32_f16(false, a0, false, b1, (short)0, c01, false, false);
    c10 = __builtin_amdgcn_wmma_f32_16x16x32_f16(false, a1, false, b0, (short)0, c10, false, false);
    c11 = __builtin_amdgcn_wmma_f32_16x16x32_f16(false, a1, false, b1, (short)0, c11, false, false);
  }
  // C/D layout: element r of lane -> row m0 + hi*8 + r, col n0 + lo16
  int mr = m0 + hi * 8;
  int nc = n0 + lo16;
  bool w0 = nc < Nreal, w1 = (nc + 16) < Nreal;
#pragma unroll
  for (int r = 0; r < 8; ++r) {
    int mm = mr + r;
    if (w0) C[(size_t)mm * ldc + nc] = c00[r];
    if (w1) C[(size_t)mm * ldc + nc + 16] = c01[r];
    if (w0) C[(size_t)(mm + 16) * ldc + nc] = c10[r];
    if (w1) C[(size_t)(mm + 16) * ldc + nc + 16] = c11[r];
  }
}

// ---------------- LIF after conv2 (re-layout to fc1 input) ----------------
__global__ void lif1_conv2(const float* __restrict__ c2out, float* __restrict__ v1,
                           float* __restrict__ i1, _Float16* __restrict__ zfc1) {
  int idx = blockIdx.x * blockDim.x + threadIdx.x;
  if (idx >= S1) return;
  int pos = idx % 121; int t = idx / 121;
  int c = t % 50; int b = t / 50;
  float inp = c2out[(size_t)(b * 121 + pos) * N2P + c];
  float v = v1[idx], cur = i1[idx];
  float vdec = v + 0.1f * (cur - v);
  float z = (vdec - 0.2f) > 0.f ? 1.f : 0.f;
  v1[idx] = (1.f - z) * vdec;
  i1[idx] = 0.8f * cur + inp;
  zfc1[(size_t)b * KF1P + c * 121 + pos] = (_Float16)z;
}

// ---------------- LIF for fc layers, f16 spike out with row pitch ----------------
__global__ void lif_fc(const float* __restrict__ inp, float* __restrict__ v,
                       float* __restrict__ i, _Float16* __restrict__ zout,
                       int n, int krow, int ldz, float th) {
  int idx = blockIdx.x * blockDim.x + threadIdx.x;
  if (idx >= n) return;
  int col = idx % krow; int b = idx / krow;
  float vv = v[idx], cur = i[idx];
  float vdec = vv + 0.1f * (cur - vv);
  float z = (vdec - th) > 0.f ? 1.f : 0.f;
  v[idx] = (1.f - z) * vdec;
  i[idx] = 0.8f * cur + inp[idx];
  zout[(size_t)b * ldz + col] = (_Float16)z;
}

// ---------------- LIF3 (f32 spikes, feeds scalar fc_out) ----------------
__global__ void lif3_f32(const float* __restrict__ inp, float* __restrict__ v,
                         float* __restrict__ i, float* __restrict__ zout, int n, float th) {
  int idx = blockIdx.x * blockDim.x + threadIdx.x;
  if (idx >= n) return;
  float vv = v[idx], cur = i[idx];
  float vdec = vv + 0.1f * (cur - vv);
  float z = (vdec - th) > 0.f ? 1.f : 0.f;
  v[idx] = (1.f - z) * vdec;
  i[idx] = 0.8f * cur + inp[idx];
  zout[idx] = z;
}

// ---------------- fc_out + LI output cell (fused) ----------------
__global__ void fcout_li(const float* __restrict__ z3, const float* __restrict__ wfo,
                         float* __restrict__ vo, float* __restrict__ io,
                         float* __restrict__ out_t) {
  int idx = blockIdx.x * blockDim.x + threadIdx.x;
  if (idx >= BATCH * 10) return;
  int o = idx % 10; int b = idx / 10;
  const float* zr = z3 + (size_t)b * 200;
  const float* wr = wfo + o * 200;
  float acc = 0.f;
  for (int k = 0; k < 200; ++k) acc += zr[k] * wr[k];
  float vv = vo[idx], cur = io[idx];
  float vdec = vv + 0.1f * (cur - vv);
  out_t[idx] = vdec;
  vo[idx] = vdec;
  io[idx] = 0.8f * cur + acc;
}

// ---------------- host ----------------
static inline char* wsalloc(char*& p, size_t bytes) {
  char* r = p;
  p += (bytes + 255) & ~(size_t)255;
  return r;
}

extern "C" void kernel_launch(void* const* d_in, const int* in_sizes, int n_in,
                              void* d_out, int out_size, void* d_ws, size_t ws_size,
                              hipStream_t stream) {
  (void)in_sizes; (void)n_in; (void)out_size; (void)ws_size;
  const float* x    = (const float*)d_in[0];   // [40,256,1,28,28]
  const float* wc1  = (const float*)d_in[1];   // [20,1,3,3]
  const float* wc2  = (const float*)d_in[2];   // [50,20,3,3] = [50,180]
  const float* wf1  = (const float*)d_in[3];   // [500,6050]
  const float* wf2  = (const float*)d_in[4];   // [200,500]
  const float* wfo  = (const float*)d_in[5];   // [10,200]
  float* out = (float*)d_out;                  // [40,256,10]

  // carve workspace
  char* p = (char*)d_ws;
  float*    v0     = (float*)   wsalloc(p, (size_t)S0 * 4);
  float*    i0     = (float*)   wsalloc(p, (size_t)S0 * 4);
  float*    z0     = (float*)   wsalloc(p, (size_t)S0 * 4);
  _Float16* ph     = (_Float16*)wsalloc(p, (size_t)SP * 2);
  _Float16* col    = (_Float16*)wsalloc(p, (size_t)M2 * K2P * 2);
  float*    c2out  = (float*)   wsalloc(p, (size_t)M2 * N2P * 4);
  float*    v1     = (float*)   wsalloc(p, (size_t)S1 * 4);
  float*    i1     = (float*)   wsalloc(p, (size_t)S1 * 4);
  _Float16* zfc1   = (_Float16*)wsalloc(p, (size_t)BATCH * KF1P * 2);
  float*    fc1out = (float*)   wsalloc(p, (size_t)BATCH * NF1 * 4);
  float*    v2     = (float*)   wsalloc(p, (size_t)BATCH * NF1 * 4);
  float*    i2     = (float*)   wsalloc(p, (size_t)BATCH * NF1 * 4);
  _Float16* z2     = (_Float16*)wsalloc(p, (size_t)BATCH * KF2P * 2);
  float*    fc2out = (float*)   wsalloc(p, (size_t)BATCH * NF2 * 4);
  float*    v3     = (float*)   wsalloc(p, (size_t)BATCH * NF2 * 4);
  float*    i3     = (float*)   wsalloc(p, (size_t)BATCH * NF2 * 4);
  float*    z3     = (float*)   wsalloc(p, (size_t)BATCH * NF2 * 4);
  float*    vo     = (float*)   wsalloc(p, (size_t)BATCH * 10 * 4);
  float*    io     = (float*)   wsalloc(p, (size_t)BATCH * 10 * 4);
  _Float16* wc2h   = (_Float16*)wsalloc(p, (size_t)N2P * K2P * 2);
  _Float16* wf1h   = (_Float16*)wsalloc(p, (size_t)NF1P * KF1P * 2);
  _Float16* wf2h   = (_Float16*)wsalloc(p, (size_t)NF2P * KF2P * 2);
  size_t used = (size_t)(p - (char*)d_ws);

  // zero everything (states reset, padded weight/spike buffers zeroed)
  zero_ws<<<4096, 256, 0, stream>>>((float*)d_ws, used / 4);

  // convert weights to f16 padded layouts
  { int n = 50 * 180;    cvt_w_f16<<<(n + 255) / 256, 256, 0, stream>>>(wc2, wc2h, n, 180, K2P); }
  { int n = NF1 * KF1;   cvt_w_f16<<<(n + 255) / 256, 256, 0, stream>>>(wf1, wf1h, n, KF1, KF1P); }
  { int n = NF2 * 500;   cvt_w_f16<<<(n + 255) / 256, 256, 0, stream>>>(wf2, wf2h, n, 500, KF2P); }

  for (int t = 0; t < TSTEPS; ++t) {
    const float* xt = x + (size_t)t * BATCH * 784;
    float* out_t = out + (size_t)t * BATCH * 10;

    conv1_lif0<<<(S0 + 255) / 256, 256, 0, stream>>>(xt, wc1, v0, i0, z0);
    pool2<<<(SP + 255) / 256, 256, 0, stream>>>(z0, ph);
    im2col2<<<(M2 * K2P + 255) / 256, 256, 0, stream>>>(ph, col);

    // conv2 implicit GEMM: [30976 x 192] * [64 x 192]^T -> [30976 x 64]
    {
      int waves = (M2 / 32) * (N2P / 32);
      gemm_wmma_2x2<<<(waves * 32) / 256, 256, 0, stream>>>(
          col, wc2h, c2out, M2, 50, N2P, K2P, K2P, K2P, N2P / 32);
    }
    lif1_conv2<<<(S1 + 255) / 256, 256, 0, stream>>>(c2out, v1, i1, zfc1);

    // fc1 GEMM: [256 x 6080] * [512 x 6080]^T -> [256 x 500]
    {
      int waves = (BATCH / 32) * (NF1P / 32);
      gemm_wmma_2x2<<<(waves * 32) / 256, 256, 0, stream>>>(
          zfc1, wf1h, fc1out, BATCH, NF1, NF1, KF1P, KF1P, KF1P, NF1P / 32);
    }
    lif_fc<<<(BATCH * NF1 + 255) / 256, 256, 0, stream>>>(
        fc1out, v2, i2, z2, BATCH * NF1, NF1, KF2P, 0.1f);

    // fc2 GEMM: [256 x 512] * [224 x 512]^T -> [256 x 200]
    {
      int waves = (BATCH / 32) * (NF2P / 32);
      gemm_wmma_2x2<<<(waves * 32) / 256, 256, 0, stream>>>(
          z2, wf2h, fc2out, BATCH, NF2, NF2, KF2P, KF2P, KF2P, NF2P / 32);
    }
    lif3_f32<<<(BATCH * NF2 + 255) / 256, 256, 0, stream>>>(
        fc2out, v3, i3, z3, BATCH * NF2, 0.05f);

    fcout_li<<<(BATCH * 10 + 255) / 256, 256, 0, stream>>>(z3, wfo, vo, io, out_t);
  }
}
